// FP8ActivationResidualQwen2DecoderLayer_64261300683339
// MI455X (gfx1250) — 8-bit matrix operand order repaired and checked by reading, compiled, not run on hardware
//
#include <hip/hip_runtime.h>
#include <hip/hip_bf16.h>
#include <math.h>


__device__ __forceinline__ unsigned int sw_f32_to_fp8_1(float f) {
  unsigned int u = __float_as_uint(f);
  unsigned int s = (u >> 31) << 7;
  float a = fabsf(f);
  if (!(a == a)) return s | 0x7fu;
  if (a >= 448.0f) return s | 0x7eu;
  if (a == 0.0f) return s;
  int e = (int)((u >> 23) & 0xff) - 127;
  if (e < -6) {
    int qi = (int)rintf(a * 512.0f);
    if (qi >= 8) return s | 0x08u;
    return s | (unsigned int)qi;
  }
  float m = __uint_as_float((u & 0x7fffffu) | 0x3f800000u);
  int mi = (int)rintf(m * 8.0f) - 8;
  if (mi == 8) { mi = 0; ++e; }
  if (e > 8) return s | 0x7eu;
  return s | ((unsigned int)(e + 7) << 3) | (unsigned int)mi;
}

__device__ __forceinline__ float sw_fp8_to_f32(unsigned int b) {
  unsigned int sg = (b >> 7) & 1u, e = (b >> 3) & 0xfu, m = b & 7u;
  float v;
  if (e == 0)                  v = (float)m * (1.0f / 512.0f);
  else if (e == 15 && m == 7)  v = __uint_as_float(0x7fc00000u);
  else                         v = __uint_as_float(((e + 120u) << 23) | (m << 20));
  return sg ? -v : v;
}

__device__ __forceinline__ unsigned int pack4_fp8(float a, float b, float c, float d) {
#if __has_builtin(__builtin_amdgcn_cvt_pk_fp8_f32)
  int p = __builtin_amdgcn_cvt_pk_fp8_f32(a, b, 0, false);
  p = __builtin_amdgcn_cvt_pk_fp8_f32(c, d, p, true);
  return (unsigned int)p;
#else
  return sw_f32_to_fp8_1(a) | (sw_f32_to_fp8_1(b) << 8) |
         (sw_f32_to_fp8_1(c) << 16) | (sw_f32_to_fp8_1(d) << 24);
#endif
}

template <int SEL>
__device__ __forceinline__ float fp8_byte_to_f32(unsigned int pk) {
#if __has_builtin(__builtin_amdgcn_cvt_f32_fp8)
  return __builtin_amdgcn_cvt_f32_fp8((int)pk, SEL);
#else
  return sw_fp8_to_f32((pk >> (8 * SEL)) & 0xffu);
#endif
}

__global__ void init_ws_kernel(unsigned int* ws_u) {
  if (threadIdx.x == 0) { ws_u[0] = 0u; ws_u[1] = 0u; }
}

__global__ __launch_bounds__(256) void maxabs_kernel(const float* __restrict__ x,
                                                     long n4,
                                                     unsigned int* __restrict__ out) {
  __shared__ float smax[256];
  const float4* x4 = (const float4*)x;
  long i = (long)blockIdx.x * blockDim.x + threadIdx.x;
  long stride = (long)gridDim.x * blockDim.x;
  float m = 0.0f;
  for (; i < n4; i += stride) {
    float4 v = x4[i];
    m = fmaxf(m, fmaxf(fmaxf(fabsf(v.x), fabsf(v.y)),
                       fmaxf(fabsf(v.z), fabsf(v.w))));
  }
  smax[threadIdx.x] = m;
  __syncthreads();
  for (int s = 128; s > 0; s >>= 1) {
    if ((int)threadIdx.x < s) smax[threadIdx.x] = fmaxf(smax[threadIdx.x], smax[threadIdx.x + s]);
    __syncthreads();
  }
  if (threadIdx.x == 0) atomicMax(out, __float_as_uint(smax[0]));
}

__global__ void finalize_scales_kernel(unsigned int* ws_u) {
  if (threadIdx.x == 0) {
    float sf = fmaxf(__uint_as_float(ws_u[0]) * (1.0f / 448.0f), 1e-12f);
    float sw = fmaxf(__uint_as_float(ws_u[1]) * (1.0f / 448.0f), 1e-12f);
    float* f = (float*)ws_u;
    f[2] = sf; f[3] = sw;
    f[4] = 1.0f / sf; f[5] = 1.0f / sw;
    f[6] = sf * sw;
  }
}

__global__ __launch_bounds__(256) void quantize_fp8_kernel(const float* __restrict__ x,
                                                           unsigned int* __restrict__ q,
                                                           long n4,
                                                           const float* __restrict__ rs_ptr) {
  const float rs = *rs_ptr;
  const float4* x4 = (const float4*)x;
  long i = (long)blockIdx.x * blockDim.x + threadIdx.x;
  long stride = (long)gridDim.x * blockDim.x;
  for (; i < n4; i += stride) {
    float4 v = x4[i];
    q[i] = pack4_fp8(v.x * rs, v.y * rs, v.z * rs, v.w * rs);
  }
}

typedef __attribute__((ext_vector_type(16))) int   v16i;
typedef __attribute__((ext_vector_type(8)))  float v8f;

#define LDS_ROW_STRIDE 144u
#define LDS_A_BYTES    (128u * LDS_ROW_STRIDE)
#define LDS_B_BYTES    (64u * LDS_ROW_STRIDE)
#define LDS_BUF_BYTES  (LDS_A_BYTES + LDS_B_BYTES)

__global__ __launch_bounds__(256) void gemm_fp8_wmma_kernel(
    const unsigned char* __restrict__ Aq,
    const unsigned char* __restrict__ Bq,
    const float* __restrict__ re_qx,
    const float* __restrict__ re_sx,
    const float* __restrict__ ws_f,
    float* __restrict__ fp_x,
    int T, int D) {
  __shared__ __align__(16) unsigned char smem[2 * LDS_BUF_BYTES];

  const int tid  = threadIdx.x;
  const int lane = tid & 31;
  const int wave = tid >> 5;
  const int wm = wave & 3;
  const int wn = wave >> 2;
  const int rowBase = blockIdx.x * 128;
  const int colBase = blockIdx.y * 64;
  const int m0 = wm * 32;
  const int n0 = wn * 32;
  const int mlo = lane & 15;
  const int hi  = lane >> 4;
  const int K = D;

  const unsigned ldsBase = (unsigned)(size_t)(&smem[0]);

  auto issue_slab = [&](unsigned bufOff, int k0) {
#pragma unroll
    for (int j = 0; j < 4; ++j) {
      int c = tid + j * 256;
      int row = c >> 3, cc = c & 7;
      unsigned lds = ldsBase + bufOff + (unsigned)row * LDS_ROW_STRIDE + (unsigned)(cc * 16);
      const unsigned char* g = Aq + (size_t)(rowBase + row) * K + (size_t)(k0 + cc * 16);
      asm volatile("global_load_async_to_lds_b128 %0, %1, off"
                   :: "v"(lds), "v"(g) : "memory");
    }
#pragma unroll
    for (int j = 0; j < 2; ++j) {
      int c = tid + j * 256;
      int row = c >> 3, cc = c & 7;
      unsigned lds = ldsBase + bufOff + LDS_A_BYTES + (unsigned)row * LDS_ROW_STRIDE + (unsigned)(cc * 16);
      const unsigned char* g = Bq + (size_t)(colBase + row) * K + (size_t)(k0 + cc * 16);
      asm volatile("global_load_async_to_lds_b128 %0, %1, off"
                   :: "v"(lds), "v"(g) : "memory");
    }
  };

  v8f acc[2][2] = {};
  const int nIter = K / 128;

  issue_slab(0u, 0);
  for (int i = 0; i < nIter; ++i) {
    const unsigned bufOff = (unsigned)(i & 1) * LDS_BUF_BYTES;
    if (i + 1 < nIter) {
      issue_slab((unsigned)((i + 1) & 1) * LDS_BUF_BYTES, (i + 1) * 128);
      asm volatile("s_wait_asynccnt 0x6" ::: "memory");
    } else {
      asm volatile("s_wait_asynccnt 0x0" ::: "memory");
    }
    __syncthreads();

    v16i a[2], b[2];
#pragma unroll
    for (int mt = 0; mt < 2; ++mt) {
      const unsigned char* arow =
          smem + bufOff + (unsigned)(m0 + mt * 16 + mlo) * LDS_ROW_STRIDE + (unsigned)(hi * 8);
#pragma unroll
      for (int h = 0; h < 2; ++h)
#pragma unroll
        for (int p = 0; p < 4; ++p) {
          int2 d = *(const int2*)(arow + h * 64 + p * 16);
          a[mt][h * 8 + p * 2 + 0] = d.x;
          a[mt][h * 8 + p * 2 + 1] = d.y;
        }
    }
#pragma unroll
    for (int nt = 0; nt < 2; ++nt) {
      const unsigned char* brow =
          smem + bufOff + LDS_A_BYTES + (unsigned)(n0 + nt * 16 + mlo) * LDS_ROW_STRIDE + (unsigned)(hi * 8);
#pragma unroll
      for (int g = 0; g < 8; ++g) {
        int2 d = *(const int2*)(brow + g * 16);
        b[nt][g * 2 + 0] = d.x;
        b[nt][g * 2 + 1] = d.y;
      }
    }
#pragma unroll
    for (int mt = 0; mt < 2; ++mt)
#pragma unroll
      for (int nt = 0; nt < 2; ++nt)
        acc[mt][nt] = __builtin_amdgcn_wmma_f32_16x16x128_fp8_fp8(
            a[mt], b[nt], (short)0, acc[mt][nt], false, false);

    __syncthreads();
  }

  const float ssw = ws_f[6];
  const int sxStride = D >> 4;
#pragma unroll
  for (int mt = 0; mt < 2; ++mt) {
#pragma unroll
    for (int nt = 0; nt < 2; ++nt) {
      const int col = colBase + n0 + nt * 16 + mlo;
#pragma unroll
      for (int r = 0; r < 8; ++r) {
        const int row = rowBase + m0 + mt * 16 + r + 8 * hi;
        const size_t idx = (size_t)row * D + col;
        const float fc4 = acc[mt][nt][r] * ssw;
        const float res = re_qx[idx] * re_sx[(size_t)row * sxStride + (col >> 4)];
        fp_x[idx] = res + fc4;
      }
    }
  }
}

__global__ __launch_bounds__(256) void group_quant_kernel(const float* __restrict__ fp_x,
                                                          float* __restrict__ out_x,
                                                          float* __restrict__ out_s,
                                                          long ngroups) {
  long g = (long)blockIdx.x * blockDim.x + threadIdx.x;
  if (g >= ngroups) return;
  const float4* src = (const float4*)(fp_x + g * 16);
  float4 v[4];
  float m = 0.0f;
#pragma unroll
  for (int i = 0; i < 4; ++i) {
    v[i] = src[i];
    m = fmaxf(m, fmaxf(fmaxf(fabsf(v[i].x), fabsf(v[i].y)),
                       fmaxf(fabsf(v[i].z), fabsf(v[i].w))));
  }
  const float s = fmaxf(m * (1.0f / 448.0f), 1e-12f);
  const float rs = 1.0f / s;
  float4* dst = (float4*)(out_x + g * 16);
#pragma unroll
  for (int i = 0; i < 4; ++i) {
    unsigned int pk = pack4_fp8(v[i].x * rs, v[i].y * rs, v[i].z * rs, v[i].w * rs);
    float4 o;
    o.x = fp8_byte_to_f32<0>(pk);
    o.y = fp8_byte_to_f32<1>(pk);
    o.z = fp8_byte_to_f32<2>(pk);
    o.w = fp8_byte_to_f32<3>(pk);
    dst[i] = o;
  }
  out_s[g] = s;
}

extern "C" void kernel_launch(void* const* d_in, const int* in_sizes, int n_in,
                              void* d_out, int out_size, void* d_ws, size_t ws_size,
                              hipStream_t stream) {
  const float* re_qx   = (const float*)d_in[0];
  const float* re_sx   = (const float*)d_in[1];
  const float* flash_x = (const float*)d_in[2];
  const float* weight  = (const float*)d_in[3];

  const int D = (int)lround(sqrt((double)in_sizes[3]));
  const int T = (int)((long)in_sizes[2] / D);
  const long TD = (long)T * D;
  const long DD = (long)D * D;

  unsigned int* ws_u = (unsigned int*)d_ws;
  float* ws_f = (float*)d_ws;
  unsigned char* Aq = (unsigned char*)d_ws + 256;
  unsigned char* Bq = Aq + TD;

  float* fp_x  = (float*)d_out;
  float* out_x = fp_x + TD;
  float* out_s = out_x + TD;

  init_ws_kernel<<<1, 1, 0, stream>>>(ws_u);
  maxabs_kernel<<<1024, 256, 0, stream>>>(flash_x, TD / 4, ws_u + 0);
  maxabs_kernel<<<512, 256, 0, stream>>>(weight, DD / 4, ws_u + 1);
  finalize_scales_kernel<<<1, 1, 0, stream>>>(ws_u);
  quantize_fp8_kernel<<<2048, 256, 0, stream>>>(flash_x, (unsigned int*)Aq, TD / 4, ws_f + 4);
  quantize_fp8_kernel<<<1024, 256, 0, stream>>>(weight, (unsigned int*)Bq, DD / 4, ws_f + 5);

  dim3 ggrid(T / 128, D / 64);
  gemm_fp8_wmma_kernel<<<ggrid, 256, 0, stream>>>(Aq, Bq, re_qx, re_sx, ws_f,
                                                  fp_x, T, D);

  const long ngroups = TD / 16;
  group_quant_kernel<<<(unsigned)((ngroups + 255) / 256), 256, 0, stream>>>(
      fp_x, out_x, out_s, ngroups);
}
